// DeformableCausalConv1d_74002286510621
// MI455X (gfx1250) — matrix operand order repaired and checked by reading, not compiled, not run on hardware
//
#include <hip/hip_runtime.h>
#include <hip/hip_bf16.h>
#include <math.h>

// Problem constants (match reference)
#define BB   4
#define CIN  512
#define COUT 512
#define TT   4096
#define KW   8
#define OKW  3
#define KD   (CIN * KW)     // 4096 : GEMM reduction dim (c,k)
#define NTOT (BB * TT)      // 16384: GEMM N dim (b,t)
#define PADL (KW - 1)       // 7
#define PP   (TT + PADL)    // 4103 : padded length

// Tiling
#define WGM  128            // workgroup M tile
#define WGN  128            // workgroup N tile
#define KC   32             // K chunk per WMMA step
#define LDSS 40             // LDS row stride in bf16 (32 data + 8 pad) => 80B rows, 16B aligned, conflict-free

typedef __attribute__((ext_vector_type(8)))  __bf16 v8bf;
typedef __attribute__((ext_vector_type(16))) __bf16 v16bf;
typedef __attribute__((ext_vector_type(8)))  float  v8f;

static __device__ __forceinline__ unsigned short f32_to_bf16(float f) {
  union { float f; unsigned u; } v; v.f = f;
  unsigned r = v.u + 0x7FFFu + ((v.u >> 16) & 1u);   // round-to-nearest-even
  return (unsigned short)(r >> 16);
}
static __device__ __forceinline__ float bf16_to_f32(unsigned short h) {
  union { unsigned u; float f; } v; v.u = ((unsigned)h) << 16;
  return v.f;
}
static __device__ __forceinline__ v16bf cat8(v8bf a, v8bf b) {
  return __builtin_shufflevector(a, b, 0,1,2,3,4,5,6,7,8,9,10,11,12,13,14,15);
}
static __device__ __forceinline__ v8f wmma_bf16(v16bf a, v16bf b, v8f c) {
  // (neg_a, A, neg_b, B, c_mod, C, reuse_a, reuse_b)
  return __builtin_amdgcn_wmma_f32_16x16x32_bf16(false, a, false, b, (short)0, c, false, false);
}

// ---------------------------------------------------------------------------
// Kernel 1: split f32 weight [COUT][CIN][KW] (flat kd = c*8+k) into bf16 hi/lo
// ---------------------------------------------------------------------------
__global__ __launch_bounds__(256)
void split_weight_kernel(const float* __restrict__ w,
                         unsigned short* __restrict__ whi,
                         unsigned short* __restrict__ wlo) {
  int i = blockIdx.x * blockDim.x + threadIdx.x;
  if (i < COUT * KD) {
    float v = w[i];
    unsigned short h = f32_to_bf16(v);
    whi[i] = h;
    wlo[i] = f32_to_bf16(v - bf16_to_f32(h));
  }
}

// ---------------------------------------------------------------------------
// Kernel 2: fused deformable sampling + bf16x3 WMMA GEMM
//   D[o, n] = sum_kd W[o, kd] * S[kd, n] + bias[o],  n = b*T + t
// 256 threads = 8 waves arranged 2(M) x 4(N); per-wave 4x2 tiles of 16x16.
// ---------------------------------------------------------------------------
__global__ __launch_bounds__(256)
void deform_gemm_kernel(const float* __restrict__ x,        // [B][CIN][T]
                        const float* __restrict__ ow,       // [KD][3]
                        const float* __restrict__ ob,       // [KD]
                        const unsigned short* __restrict__ whi, // [COUT][KD] bf16
                        const unsigned short* __restrict__ wlo, // [COUT][KD] bf16
                        const float* __restrict__ bias,     // [COUT]
                        float* __restrict__ out) {          // [B][COUT][T]
  __shared__ unsigned short sBhi[WGN * LDSS];
  __shared__ unsigned short sBlo[WGN * LDSS];

  const int tid      = threadIdx.x;
  const int lane     = tid & 31;
  const int wave     = tid >> 5;
  const int waveM    = wave >> 2;       // 0..1
  const int waveN    = wave & 3;        // 0..3
  const int laneMod  = lane & 15;
  const int laneHalf = lane >> 4;

  const int blockN = blockIdx.x * WGN;  // over NTOT
  const int blockM = blockIdx.y * WGM;  // over COUT

  // --- sampling work distribution: each thread fills 16 of the 32x128 B-tile ---
  const int nn     = tid & (WGN - 1);       // 0..127  (column within tile)
  const int kkBase = (tid >> 7) * 16;       // 0 or 16 (row range within K-chunk)
  const int n_g    = blockN + nn;
  const int bb     = n_g >> 12;             // T = 4096 = 2^12
  const int t_s    = n_g & (TT - 1);

  v8f acc[4][2];
  const v8f vzero = {0.f,0.f,0.f,0.f,0.f,0.f,0.f,0.f};
#pragma unroll
  for (int mt = 0; mt < 4; ++mt)
#pragma unroll
    for (int nt = 0; nt < 2; ++nt) acc[mt][nt] = vzero;

  const int aRow = blockM + waveM * 64 + laneMod;   // A-frag base row (per-lane, +16*mt)

  for (int kc = 0; kc < KD / KC; ++kc) {
    __syncthreads();   // protect LDS from previous chunk's readers

    // ------------------ stage B-tile: deformable sampling -> LDS bf16 hi/lo --
    {
      const int kd0 = kc * KC + kkBase;
      const float* xrow = nullptr;
      float xm2 = 0.f, xm1 = 0.f, x0 = 0.f;
#pragma unroll
      for (int i = 0; i < 16; ++i) {
        const int kd = kd0 + i;
        const int c  = kd >> 3;
        const int k  = kd & 7;
        if ((i & 7) == 0) {               // new input channel every 8 taps
          xrow = x + ((size_t)bb * CIN + c) * TT;
          xm2 = (t_s >= 2) ? xrow[t_s - 2] : 0.f;
          xm1 = (t_s >= 1) ? xrow[t_s - 1] : 0.f;
          x0  = xrow[t_s];
        }
        // causal depthwise 3-tap conv -> raw offset
        float raw = ow[kd * 3 + 0] * xm2 + ow[kd * 3 + 1] * xm1 +
                    ow[kd * 3 + 2] * x0 + ob[kd];
        // strict causal: offsets = -|raw|; pos in padded coords = t + k + off
        float pos = (float)(t_s + k) - fabsf(raw);
        float rp  = rintf(pos);                       // round-half-even
        if (fabsf(pos - rp) < 1e-6f) pos = rp;        // snap near-integers
        pos = fminf(fmaxf(pos, 0.f), (float)(PP - 1));
        float fidx  = floorf(pos);
        float alpha = pos - fidx;
        int i0 = (int)fidx;
        int i1 = i0 + 1; if (i1 > PP - 1) i1 = PP - 1;
        float s0 = (i0 >= PADL) ? xrow[i0 - PADL] : 0.f;
        float s1 = (i1 >= PADL) ? xrow[i1 - PADL] : 0.f;
        float s  = s0 * (1.f - alpha) + s1 * alpha;
        // split into bf16 hi + lo residual (error-compensated GEMM inputs)
        unsigned short h = f32_to_bf16(s);
        unsigned short l = f32_to_bf16(s - bf16_to_f32(h));
        sBhi[nn * LDSS + kkBase + i] = h;
        sBlo[nn * LDSS + kkBase + i] = l;
      }
    }
    __syncthreads();

    // prefetch next chunk of A into caches (global_prefetch_b8)
    if (kc + 1 < KD / KC) {
      __builtin_prefetch(whi + (size_t)aRow * KD + (size_t)(kc + 1) * KC, 0, 3);
      __builtin_prefetch(wlo + (size_t)aRow * KD + (size_t)(kc + 1) * KC, 0, 3);
    }

    // ------------------ B fragments from LDS (column = laneMod, K = laneHalf*16 + 0..15)
    v16bf bhi[2], blo[2];
#pragma unroll
    for (int nt = 0; nt < 2; ++nt) {
      const int nloc = waveN * 32 + nt * 16 + laneMod;
      const int kb   = laneHalf * 16;
      const v8bf b0h = *(const v8bf*)&sBhi[nloc * LDSS + kb];
      const v8bf b1h = *(const v8bf*)&sBhi[nloc * LDSS + kb + 8];
      const v8bf b0l = *(const v8bf*)&sBlo[nloc * LDSS + kb];
      const v8bf b1l = *(const v8bf*)&sBlo[nloc * LDSS + kb + 8];
      bhi[nt] = cat8(b0h, b1h);
      blo[nt] = cat8(b0l, b1l);
    }

    // ------------------ A fragments from global (L2-resident weights) + WMMA
#pragma unroll
    for (int mt = 0; mt < 4; ++mt) {
      const size_t aoff = (size_t)(aRow + mt * 16) * KD + (size_t)kc * KC + laneHalf * 16;
      const v8bf a0h = *(const v8bf*)&whi[aoff];
      const v8bf a1h = *(const v8bf*)&whi[aoff + 8];
      const v8bf a0l = *(const v8bf*)&wlo[aoff];
      const v8bf a1l = *(const v8bf*)&wlo[aoff + 8];
      const v16bf ahi = cat8(a0h, a1h);
      const v16bf alo = cat8(a0l, a1l);
#pragma unroll
      for (int nt = 0; nt < 2; ++nt) {
        acc[mt][nt] = wmma_bf16(ahi, bhi[nt], acc[mt][nt]);  // hi*hi
        acc[mt][nt] = wmma_bf16(ahi, blo[nt], acc[mt][nt]);  // hi*lo
        acc[mt][nt] = wmma_bf16(alo, bhi[nt], acc[mt][nt]);  // lo*hi
      }
    }
  }

  // ------------------ epilogue: D tile layout -> out[b][o][t] + bias ---------
#pragma unroll
  for (int mt = 0; mt < 4; ++mt) {
    const int oBase = blockM + waveM * 64 + mt * 16 + laneHalf * 8;
#pragma unroll
    for (int nt = 0; nt < 2; ++nt) {
      const int n = blockN + waveN * 32 + nt * 16 + laneMod;
      const int b = n >> 12;
      const int t = n & (TT - 1);
      float* op = out + ((size_t)b * COUT + oBase) * TT + t;
#pragma unroll
      for (int r = 0; r < 8; ++r) {
        op[(size_t)r * TT] = acc[mt][nt][r] + bias[oBase + r];
      }
    }
  }
}

// ---------------------------------------------------------------------------
extern "C" void kernel_launch(void* const* d_in, const int* in_sizes, int n_in,
                              void* d_out, int out_size, void* d_ws, size_t ws_size,
                              hipStream_t stream) {
  const float* x    = (const float*)d_in[0];  // [B][CIN][T]
  const float* ow   = (const float*)d_in[1];  // [CIN*K][1][3]
  const float* ob   = (const float*)d_in[2];  // [CIN*K]
  const float* w    = (const float*)d_in[3];  // [COUT][CIN][K]
  const float* bias = (const float*)d_in[4];  // [COUT]
  float* out = (float*)d_out;

  unsigned short* whi = (unsigned short*)d_ws;            // 4 MB
  unsigned short* wlo = whi + (size_t)COUT * KD;          // 4 MB

  split_weight_kernel<<<(COUT * KD + 255) / 256, 256, 0, stream>>>(w, whi, wlo);

  dim3 grid(NTOT / WGN, COUT / WGM);   // (128, 4)
  deform_gemm_kernel<<<grid, 256, 0, stream>>>(x, ow, ob, whi, wlo, bias, out);
}
